// LSTMCell_31585189495421
// MI455X (gfx1250) — matrix operand order repaired and checked by reading, not compiled, not run on hardware
//
#include <hip/hip_runtime.h>
#include <math.h>

typedef __attribute__((ext_vector_type(16))) __bf16 v16bf;
typedef __attribute__((ext_vector_type(8)))  __bf16 v8bf;
typedef __attribute__((ext_vector_type(4)))  __bf16 v4bf;
typedef __attribute__((ext_vector_type(8)))  float  v8f;

#define BATCH   16384
#define DX      256
#define DH      512
#define DS      128
#define DT      128
#define KTOT    1024   // DX + DH + DS + DT
#define GATES   2048   // 4 * DH

// Fast gate activations: single v_exp_f32 + v_rcp_f32 each, branch-free.
__device__ __forceinline__ float fast_sigmoid(float x) {
    return __builtin_amdgcn_rcpf(1.0f + __expf(-x));
}
__device__ __forceinline__ float fast_tanh(float x) {
    // (e^2x - 1)/(e^2x + 1); exp->inf saturates to +1, exp->0 gives -1.
    float e = __expf(2.0f * x);
    return 1.0f - 2.0f * __builtin_amdgcn_rcpf(e + 1.0f);
}

// ---------------------------------------------------------------------------
// Kernel 1: pack A = [x | h | s | t]  (fp32 -> bf16), row-major [B][1024]
// ---------------------------------------------------------------------------
__global__ void lstm_pack_a(const float* __restrict__ x,
                            const float* __restrict__ h,
                            const float* __restrict__ s,
                            const float* __restrict__ t,
                            __bf16* __restrict__ Apk) {
    int idx = blockIdx.x * blockDim.x + threadIdx.x;   // B*1024/4 threads
    int row = idx >> 8;                                // 256 quads per row
    int k   = (idx & 255) << 2;

    float4 v;
    if (k < DX) {
        v = *(const float4*)(x + (size_t)row * DX + k);
    } else if (k < DX + DH) {
        v = *(const float4*)(h + (size_t)row * DH + (k - DX));
    } else if (k < DX + DH + DS) {
        v = *(const float4*)(s + (size_t)row * DS + (k - DX - DH));
    } else {
        v = *(const float4*)(t + (size_t)row * DT + (k - DX - DH - DS));
    }
    v4bf o;
    o[0] = (__bf16)v.x; o[1] = (__bf16)v.y; o[2] = (__bf16)v.z; o[3] = (__bf16)v.w;
    *(v4bf*)(Apk + (size_t)row * KTOT + k) = o;
}

// ---------------------------------------------------------------------------
// Kernel 2: pack W^T = [W_x; W_h; W_s; W_t]^T  (fp32 -> bf16), [2048][1024]
// N-major so GEMM B-fragments are contiguous 8-element runs along K.
// ---------------------------------------------------------------------------
__global__ void lstm_pack_w(const float* __restrict__ Wx,
                            const float* __restrict__ Wh,
                            const float* __restrict__ Ws,
                            const float* __restrict__ Wt,
                            __bf16* __restrict__ WT) {
    int idx = blockIdx.x * blockDim.x + threadIdx.x;   // 2048*1024/4 threads
    int n   = idx >> 8;                                // 256 quads of k per column
    int k   = (idx & 255) << 2;

    const float* W; int kl;
    if (k < DX)                { W = Wx; kl = k; }
    else if (k < DX + DH)      { W = Wh; kl = k - DX; }
    else if (k < DX + DH + DS) { W = Ws; kl = k - DX - DH; }
    else                       { W = Wt; kl = k - DX - DH - DS; }

    v4bf o;
    o[0] = (__bf16)W[(size_t)(kl + 0) * GATES + n];
    o[1] = (__bf16)W[(size_t)(kl + 1) * GATES + n];
    o[2] = (__bf16)W[(size_t)(kl + 2) * GATES + n];
    o[3] = (__bf16)W[(size_t)(kl + 3) * GATES + n];
    *(v4bf*)(WT + (size_t)n * KTOT + k) = o;
}

// ---------------------------------------------------------------------------
// Kernel 3: fused GEMM (bf16 WMMA, f32 accum) + bias + gates + state update.
// Block = 256 threads = 8 waves (4 in M x 2 in N_hid).
// Block tile: 128 rows x 64 hidden cols (x 4 gates internally).
// Wave tile : 32 rows x 32 hidden cols x 4 gates = 16 accums of 16x16 f32.
// ---------------------------------------------------------------------------
__global__ void __launch_bounds__(256)
lstm_wmma_gates(const __bf16* __restrict__ Apk,   // [B][1024] bf16
                const __bf16* __restrict__ WT,    // [2048][1024] bf16
                const float*  __restrict__ bias,  // [2048]
                const float*  __restrict__ c_old, // [B][512]
                float* __restrict__ out) {        // h_new [B][512] ++ c_new [B][512]
    const int tid   = threadIdx.x;
    const int lane  = tid & 31;
    const int wid   = tid >> 5;          // 0..7
    const int waveM = wid & 3;           // 4 waves along M
    const int waveN = wid >> 2;          // 2 waves along hidden-N

    const int m0   = blockIdx.x * 128 + waveM * 32;       // row base of wave
    const int hid0 = blockIdx.y * 64  + waveN * 32;       // hidden-col base

    const int lrow  = lane & 15;
    const int lhalf = lane >> 4;         // 0 or 1

    v8f acc[4][2][2];                    // [gate][n-subtile][m-subtile]
    #pragma unroll
    for (int g = 0; g < 4; ++g)
        #pragma unroll
        for (int t = 0; t < 2; ++t)
            #pragma unroll
            for (int mt = 0; mt < 2; ++mt)
                acc[g][t][mt] = (v8f){};

    // ---- Loop-invariant fragment base pointers (constant offsets inside) ----
    const __bf16* aRow[2];
    aRow[0] = Apk + (size_t)(m0 + lrow) * KTOT + (lhalf << 3);
    aRow[1] = aRow[0] + (size_t)16 * KTOT;

    const __bf16* bp[4][2];
    #pragma unroll
    for (int g = 0; g < 4; ++g)
        #pragma unroll
        for (int t = 0; t < 2; ++t)
            bp[g][t] = WT + (size_t)(g * DH + hid0 + t * 16 + lrow) * KTOT
                          + (lhalf << 3);

    for (int kk = 0; kk < KTOT; kk += 32) {
        // A fragments: elems 0..7 = K kb..kb+7, elems 8..15 = K kb+16..kb+23
        union { v16bf v; v8bf h[2]; } au[2];
        #pragma unroll
        for (int mt = 0; mt < 2; ++mt) {
            au[mt].h[0] = *(const v8bf*)(aRow[mt] + kk);
            au[mt].h[1] = *(const v8bf*)(aRow[mt] + kk + 16);
        }

        #pragma unroll
        for (int g = 0; g < 4; ++g) {
            #pragma unroll
            for (int t = 0; t < 2; ++t) {
                union { v16bf v; v8bf h[2]; } bu;
                bu.h[0] = *(const v8bf*)(bp[g][t] + kk);
                bu.h[1] = *(const v8bf*)(bp[g][t] + kk + 16);
                #pragma unroll
                for (int mt = 0; mt < 2; ++mt) {
                    acc[g][t][mt] = __builtin_amdgcn_wmma_f32_16x16x32_bf16(
                        /*neg_a=*/false, au[mt].v,
                        /*neg_b=*/false, bu.v,
                        /*c_mod=*/(short)0, acc[g][t][mt],
                        /*reuse_a=*/false, /*reuse_b=*/false);
                }
            }
        }
    }

    // ---- Fused epilogue: bias + gates + cell/hidden update (branch-free) ----
    const size_t cplane = (size_t)BATCH * DH;   // offset of c_new in d_out
    #pragma unroll
    for (int t = 0; t < 2; ++t) {
        const int col = hid0 + t * 16 + lrow;   // hidden column 0..511
        const float bi = bias[0 * DH + col];
        const float bf = bias[1 * DH + col];
        const float bo = bias[2 * DH + col];
        const float bc = bias[3 * DH + col];
        #pragma unroll
        for (int mt = 0; mt < 2; ++mt) {
            #pragma unroll
            for (int r = 0; r < 8; ++r) {
                const int row = m0 + mt * 16 + r + (lhalf << 3);

                const float ig = fast_sigmoid(acc[0][t][mt][r] + bi);
                const float fg = fast_sigmoid(acc[1][t][mt][r] + bf);
                const float og = fast_sigmoid(acc[2][t][mt][r] + bo);
                const float cg = fast_tanh   (acc[3][t][mt][r] + bc);

                const float cold = c_old[(size_t)row * DH + col];
                const float cnew = fmaf(fg, cold, ig * cg);
                const float hnew = fast_tanh(og * cnew);   // faithful: tanh(o * c_new)

                out[(size_t)row * DH + col]          = hnew;
                out[cplane + (size_t)row * DH + col] = cnew;
            }
        }
    }
}

// ---------------------------------------------------------------------------
extern "C" void kernel_launch(void* const* d_in, const int* in_sizes, int n_in,
                              void* d_out, int out_size, void* d_ws, size_t ws_size,
                              hipStream_t stream) {
    const float* x  = (const float*)d_in[0];
    const float* h  = (const float*)d_in[1];
    const float* c  = (const float*)d_in[2];
    const float* sp = (const float*)d_in[3];
    const float* tp = (const float*)d_in[4];
    const float* Wx = (const float*)d_in[5];
    const float* Wh = (const float*)d_in[6];
    const float* bh = (const float*)d_in[7];
    const float* Ws = (const float*)d_in[8];
    const float* Wt = (const float*)d_in[9];
    float* out = (float*)d_out;

    // Workspace layout: A packed bf16 [B][1024] (32 MB), then W^T bf16 [2048][1024] (4 MB)
    __bf16* Apk = (__bf16*)d_ws;
    __bf16* WT  = (__bf16*)((char*)d_ws + (size_t)BATCH * KTOT * sizeof(__bf16));

    // Pack activations: B*1024/4 threads
    {
        int threads = 256;
        int total   = BATCH * KTOT / 4;
        lstm_pack_a<<<total / threads, threads, 0, stream>>>(x, h, sp, tp, Apk);
    }
    // Pack/transpose weights: 2048*1024/4 threads
    {
        int threads = 256;
        int total   = GATES * KTOT / 4;
        lstm_pack_w<<<total / threads, threads, 0, stream>>>(Wx, Wh, Ws, Wt, WT);
    }
    // Fused GEMM + gates
    {
        dim3 grid(BATCH / 128, DH / 64);   // 128 x 8
        lstm_wmma_gates<<<grid, 256, 0, stream>>>(Apk, WT, bh, c, out);
    }
}
